// NeighborhoodAttention_10720238371298
// MI455X (gfx1250) — matrix operand order repaired and checked by reading, not compiled, not run on hardware
//
#include <hip/hip_runtime.h>
#include <hip/hip_fp16.h>

typedef __attribute__((ext_vector_type(16))) _Float16 v16h;
typedef __attribute__((ext_vector_type(8)))  _Float16 v8h;
typedef __attribute__((ext_vector_type(8)))  float    v8f;

#define BB   4
#define HH   56
#define WW   56
#define CC   128
#define NHD  4
#define HD   32
#define KS   7
#define NB   3                      // neighborhood radius
#define M_TOTAL (BB*HH*WW)          // 12544
#define RPB_W (2*KS - 1)            // 13

// ---------------------------------------------------------------------------
// Pre-pass: f32 -> f16 copy (row-major, for the A side of the GEMMs)
// ---------------------------------------------------------------------------
__global__ void f32_to_f16(const float* __restrict__ in,
                           _Float16* __restrict__ out, int n)
{
    int i = blockIdx.x * blockDim.x + threadIdx.x;
    if (i < n) out[i] = (_Float16)in[i];
}

// ---------------------------------------------------------------------------
// Pre-pass: weight f32 [K][N] -> f16 transposed [N][K] (col-major panels so a
// B fragment becomes a contiguous 32B run per lane)
// ---------------------------------------------------------------------------
__global__ void w_to_f16_T(const float* __restrict__ w,
                           _Float16* __restrict__ wT, int K, int N)
{
    int i = blockIdx.x * blockDim.x + threadIdx.x;   // over N*K, wT row-major
    if (i < N * K) {
        int n = i / K, k = i % K;
        wT[i] = (_Float16)w[(size_t)k * N + n];
    }
}

// ---------------------------------------------------------------------------
// WMMA GEMM:  C[M x NCOLS] = A[M x 128](f16) * B[128 x NCOLS](f16,T) + bias
// Block = 8 waves = 128 rows x 16 cols of output. The 16-col x 128-K f16
// weight panel (4KB) is staged to LDS with global_load_async_to_lds_b128
// (one issue per wave), then each wave runs the K-loop with 2 global b128
// (A) + 2 ds b128 (B) + 1 v_wmma_f32_16x16x32_f16 per K-step.
// ---------------------------------------------------------------------------
template<int NCOLS, bool SCALE_Q>
__global__ void __launch_bounds__(256)
wmma_gemm_f16(const _Float16* __restrict__ A16,   // [M][128] row-major f16
              const _Float16* __restrict__ BT16,  // [NCOLS][128] f16 (transposed)
              const float* __restrict__ bias,     // [NCOLS]
              float* __restrict__ Cout,           // [M][NCOLS] f32
              float scale)
{
    constexpr int PAD = 144;                      // f16 per LDS row (128+16)
    __shared__ __align__(32) _Float16 panelT[16 * PAD];

    const int lane = threadIdx.x & 31;
    const int wave = threadIdx.x >> 5;
    const int tiles_n = NCOLS / 16;
    const int tn   = blockIdx.x % tiles_n;
    const int tm0  = (blockIdx.x / tiles_n) * 128;
    const int col0 = tn * 16;
    const int half = lane >> 4;                   // 0: lanes 0-15, 1: 16-31
    const int l16  = lane & 15;

    // ---- async stage: 16 cols x 128 K f16 panel -> LDS (16B per lane) ----
    {
        const int c  = wave * 32 + lane;          // 0..255 16B-chunk id
        const int n  = c >> 4;                    // column within panel
        const int ch = c & 15;                    // 16B chunk within 256B col
        const _Float16* gp = BT16 + (size_t)(col0 + n) * 128 + ch * 8;
        unsigned lv = (unsigned)(size_t)&panelT[n * PAD + ch * 8];
        asm volatile("global_load_async_to_lds_b128 %0, %1, off"
                     :: "v"(lv), "v"(gp) : "memory");
    }
    asm volatile("s_wait_asynccnt 0x0" ::: "memory");
    __syncthreads();

    // ---- K loop ----
    const _Float16* arow =
        A16 + (size_t)(tm0 + wave * 16 + l16) * CC + half * 16;

    v8f acc = {};
    #pragma unroll
    for (int k0 = 0; k0 < CC; k0 += 32) {
        // A fragment: element e -> K = k0 + half*16 + e
        v8h alo = *(const v8h*)(arow + k0);
        v8h ahi = *(const v8h*)(arow + k0 + 8);
        v16h av;
        #pragma unroll
        for (int e = 0; e < 8; ++e) { av[e] = alo[e]; av[e + 8] = ahi[e]; }

        // B fragment: lane = col l16, element e -> K = k0 + half*16 + e
        v16h bv = *(const v16h*)&panelT[l16 * PAD + k0 + half * 16];

        acc = __builtin_amdgcn_wmma_f32_16x16x32_f16(
            false, av, false, bv, (short)0, acc, false, false);
    }

    // ---- store: acc[r] = row (half*8+r), col l16 of wave's 16x16 tile ----
    const int col = col0 + l16;
    const float bb = bias[col];
    #pragma unroll
    for (int r = 0; r < 8; ++r) {
        const int row = tm0 + wave * 16 + half * 8 + r;
        float v = acc[r] + bb;
        if (SCALE_Q && col < CC) v *= scale;      // pre-scale Q section
        Cout[(size_t)row * NCOLS + col] = v;
    }
}

// ---------------------------------------------------------------------------
// Neighborhood attention: one thread per (batch, head, i, j) query.
// qkv rows: [q(0..127, pre-scaled) | k(128..255) | v(256..383)] fp32.
// Online softmax with fused V accumulation; writes f16 (A-side of proj GEMM).
// ---------------------------------------------------------------------------
__global__ void natten_attn(const float* __restrict__ qkv,
                            const float* __restrict__ rpb,    // [4,13,13]
                            _Float16* __restrict__ out)       // [M,128] f16
{
    const int total = BB * NHD * HH * WW;
    const int idx = blockIdx.x * blockDim.x + threadIdx.x;
    if (idx >= total) return;

    const int j = idx % WW;
    const int i = (idx / WW) % HH;
    const int h = (idx / (WW * HH)) % NHD;
    const int b = idx / (WW * HH * NHD);

    const int row = (b * HH + i) * WW + j;
    const float* qp = qkv + (size_t)row * (3 * CC) + h * HD;

    float q[HD];
    #pragma unroll
    for (int d = 0; d < HD; ++d) q[d] = qp[d];

    const int si = min(max(i - NB, 0), HH - KS);
    const int sj = min(max(j - NB, 0), WW - KS);

    const float* rp = rpb + h * (RPB_W * RPB_W)
                          + (si - i + KS - 1) * RPB_W
                          + (sj - j + KS - 1);

    float m = -3.4e38f;
    float l = 0.f;
    float acc[HD];
    #pragma unroll
    for (int d = 0; d < HD; ++d) acc[d] = 0.f;

    for (int p = 0; p < KS; ++p) {
        const int ki = si + p;
        const size_t rowbase = (size_t)((b * HH + ki) * WW + sj) * (3 * CC);
        for (int qq = 0; qq < KS; ++qq) {
            const float* kp = qkv + rowbase + (size_t)qq * (3 * CC) + CC + h * HD;
            const float* vp = kp + CC;

            float s = rp[p * RPB_W + qq];
            #pragma unroll
            for (int d = 0; d < HD; ++d) s += q[d] * kp[d];

            const float mn   = fmaxf(m, s);
            const float cfac = __expf(m - mn);
            const float pfac = __expf(s - mn);
            l = l * cfac + pfac;
            #pragma unroll
            for (int d = 0; d < HD; ++d)
                acc[d] = acc[d] * cfac + pfac * vp[d];
            m = mn;
        }
    }

    const float inv = 1.f / l;
    _Float16* op = out + (size_t)row * CC + h * HD;
    #pragma unroll
    for (int d = 0; d < HD; ++d) op[d] = (_Float16)(acc[d] * inv);
}

// ---------------------------------------------------------------------------
extern "C" void kernel_launch(void* const* d_in, const int* in_sizes, int n_in,
                              void* d_out, int out_size, void* d_ws, size_t ws_size,
                              hipStream_t stream)
{
    const float* x      = (const float*)d_in[0];
    const float* w_qkv  = (const float*)d_in[1];
    const float* b_qkv  = (const float*)d_in[2];
    const float* rpb    = (const float*)d_in[3];
    const float* w_proj = (const float*)d_in[4];
    const float* b_proj = (const float*)d_in[5];
    float* out = (float*)d_out;

    // ---- workspace layout (all offsets 128B-aligned) ----
    char* ws = (char*)d_ws;
    float*    qkv    = (float*)ws;                               // [12544,384] f32
    size_t off = (size_t)M_TOTAL * (3 * CC) * sizeof(float);     // 19267584
    _Float16* x16    = (_Float16*)(ws + off);                    // [12544,128] f16
    off += (size_t)M_TOTAL * CC * sizeof(_Float16);              // +3211264
    _Float16* attn16 = (_Float16*)(ws + off);                    // [12544,128] f16
    off += (size_t)M_TOTAL * CC * sizeof(_Float16);              // +3211264
    _Float16* wTq    = (_Float16*)(ws + off);                    // [384,128] f16
    off += (size_t)(3 * CC) * CC * sizeof(_Float16);             // +98304
    _Float16* wTp    = (_Float16*)(ws + off);                    // [128,128] f16

    const float scale = 0.17677669529663687f;                    // 32^-0.5

    // ---- pre-pass conversions ----
    {
        const int n = M_TOTAL * CC;                              // 1605632
        f32_to_f16<<<(n + 255) / 256, 256, 0, stream>>>(x, x16, n);
        w_to_f16_T<<<((3 * CC) * CC + 255) / 256, 256, 0, stream>>>(
            w_qkv, wTq, CC, 3 * CC);
        w_to_f16_T<<<(CC * CC + 255) / 256, 256, 0, stream>>>(
            w_proj, wTp, CC, CC);
    }

    // ---- phase 1: QKV projection (98 row-blocks x 24 col-tiles) ----
    {
        const int blocks = (M_TOTAL / 128) * ((3 * CC) / 16);    // 2352
        wmma_gemm_f16<3 * CC, true><<<blocks, 256, 0, stream>>>(
            x16, wTq, b_qkv, qkv, scale);
    }

    // ---- phase 2: neighborhood attention (50176 query-head threads) ----
    {
        const int total = BB * NHD * HH * WW;
        natten_attn<<<(total + 255) / 256, 256, 0, stream>>>(qkv, rpb, attn16);
    }

    // ---- phase 3: output projection (98 x 8) ----
    {
        const int blocks = (M_TOTAL / 128) * (CC / 16);          // 784
        wmma_gemm_f16<CC, false><<<blocks, 256, 0, stream>>>(
            attn16, wTp, b_proj, out, 1.0f);
    }
}
